// Attend_20083267076099
// MI455X (gfx1250) — matrix operand order repaired and checked by reading, not compiled, not run on hardware
//
#include <hip/hip_runtime.h>
#include <math.h>

typedef __attribute__((ext_vector_type(16))) _Float16 v16h;
typedef __attribute__((ext_vector_type(8)))  _Float16 v8h;
typedef __attribute__((ext_vector_type(4)))  _Float16 v4h;
typedef __attribute__((ext_vector_type(8)))  float    v8f;
typedef __attribute__((ext_vector_type(4)))  float    v4f;

#define SEQ   2048
#define DH    64
#define NH    8
#define NB    2
#define JC    128     // K/V columns staged in LDS per chunk == softmax step size
#define WAVES 8       // waves per block; block covers WAVES*16 query rows

#if __has_builtin(__builtin_amdgcn_exp2f)
#define EXP2F(x) __builtin_amdgcn_exp2f(x)
#else
#define EXP2F(x) __expf((x) * 0.6931471805599453f)
#endif
#define LOG2E   1.4426950408889634f
#define NEGHUGE -1.0e30f

#define PST_STRIDE (JC + 8)                       // 136 f16 per P^T row
#define PST_BYTES  (WAVES * 16 * PST_STRIDE * 2)  // 34816 (aliases K region)
#define VT_BYTES   (DH * (JC + 8) * 2)            // 17408
#define MASK_ALL1  0x0101010101010101ULL

// Transposed formulation: S^T = K x Q^T  (C rows = j, lanes = query row i)
//   QK:  A = K tile      [128j x 64d],  B = Q^T [64d x 16i],  C/D = S^T subtiles
//   PV:  A = V^T tile    [64d x 128j],  B = P^T [128j x 16i], C/D = O^T subtiles

__global__ __launch_bounds__(256) void attend_fa_kernel(
    const float* __restrict__ q, const float* __restrict__ kg,
    const float* __restrict__ vg, const unsigned char* __restrict__ maskb,
    const float* __restrict__ bias, float* __restrict__ out)
{
    // [0, PST_BYTES): K tile [JC][DH+8] f16 -- dead after QK phase, then P^T scratch
    // [PST_BYTES, +VT_BYTES): V^T tile [DH][JC+8] f16 (plain transpose)
    __shared__ __align__(16) char smem[PST_BYTES + VT_BYTES];
    _Float16 (*Kl)[DH + 8] = (_Float16 (*)[DH + 8])smem;
    _Float16 (*Vt)[JC + 8] = (_Float16 (*)[JC + 8])(smem + PST_BYTES);

    const int b    = blockIdx.z;
    const int h    = blockIdx.y;
    const int tid  = threadIdx.x;
    const int wave = tid >> 5;
    const int lane = tid & 31;
    const int lh   = lane >> 4;   // wave half
    const int n    = lane & 15;   // lane's matrix column (= query row index i)
    const int i0   = blockIdx.x * (WAVES * 16) + wave * 16;
    const int iq   = i0 + n;      // this lane's query row
    _Float16* Pw   = (_Float16*)smem + (size_t)wave * 16 * PST_STRIDE;

    const size_t bh    = (size_t)b * NH + h;
    const float* qrow  = q  + (bh * SEQ + (size_t)iq) * DH;
    const float* kbase = kg + (size_t)b * SEQ * DH;
    const float* vbase = vg + (size_t)b * SEQ * DH;
    const float* bbn   = bias + bh * (size_t)SEQ * SEQ + (size_t)iq * SEQ;
    const unsigned char* mb = maskb + (size_t)b * SEQ;

    // ---- Q^T B-fragments (loaded once), 1/sqrt(D) folded in ----
    v16h qf[2];
    #pragma unroll
    for (int c = 0; c < 2; ++c) {
        const float* qp = qrow + c * 32 + lh * 16;
        #pragma unroll
        for (int e = 0; e < 16; ++e)
            qf[c][e] = (_Float16)(qp[e] * 0.125f);
    }

    // All-ones A fragment for WMMA column-sum of P^T
    v16h onesh;
    #pragma unroll
    for (int e = 0; e < 16; ++e) onesh[e] = (_Float16)1.0f;

    v8f o[4] = {};                    // O^T accumulator: o[nt][r] = O[iq][nt*16+8*lh+r]
    float rmax = NEGHUGE, rsum = 0.0f; // per-lane scalar softmax state

    for (int jc0 = 0; jc0 < SEQ; jc0 += JC) {
        // ---- Bias direct into C operand (2 x b128 per subtile) + uniform mask words ----
        v8f s[8];
        bool mok[8];
        #pragma unroll
        for (int t = 0; t < 8; ++t) {
            s[t] = *(const v8f*)(bbn + jc0 + t * 16 + 8 * lh);
            mok[t] = (*(const unsigned long long*)(mb + jc0 + t * 16 + 8 * lh)) == MASK_ALL1;
        }

        __syncthreads();   // previous step's P/V consumers done
        // ---- Cooperative stage: K (f16 row-major), V (f16 plain transpose) ----
        #pragma unroll
        for (int p = 0; p < (JC * DH) / (256 * 4); ++p) {
            const int row = p * 16 + (tid >> 4);
            const int c4  = (tid & 15) * 4;
            const float4 k4 = *(const float4*)(kbase + (size_t)(jc0 + row) * DH + c4);
            const float4 v4 = *(const float4*)(vbase + (size_t)(jc0 + row) * DH + c4);
            v4h kh = { (_Float16)k4.x, (_Float16)k4.y, (_Float16)k4.z, (_Float16)k4.w };
            *(v4h*)&Kl[row][c4] = kh;
            Vt[c4 + 0][row] = (_Float16)v4.x;
            Vt[c4 + 1][row] = (_Float16)v4.y;
            Vt[c4 + 2][row] = (_Float16)v4.z;
            Vt[c4 + 3][row] = (_Float16)v4.w;
        }
        __syncthreads();

        // ---- S^T = K * Q^T + bias : 16 WMMAs (K tile as A operand) ----
        #pragma unroll
        for (int t = 0; t < 8; ++t) {
            #pragma unroll
            for (int c = 0; c < 2; ++c) {
                const _Float16* kp = &Kl[t * 16 + n][c * 32 + lh * 16];
                v8h lo = *(const v8h*)kp;
                v8h hi = *(const v8h*)(kp + 8);
                v16h ka = __builtin_shufflevector(lo, hi,
                    0,1,2,3,4,5,6,7,8,9,10,11,12,13,14,15);
                s[t] = __builtin_amdgcn_wmma_f32_16x16x32_f16(
                    false, ka, false, qf[c], (short)0, s[t], false, false);
            }
        }
        __syncthreads();   // all K-tile reads done; region becomes P^T scratch

        // ---- Mask (j is now the C-row dim; uniform fast path) ----
        const bool allm = mok[0] && mok[1] && mok[2] && mok[3] &&
                          mok[4] && mok[5] && mok[6] && mok[7];
        if (!__all(allm)) {
            #pragma unroll
            for (int t = 0; t < 8; ++t)
                #pragma unroll
                for (int r = 0; r < 8; ++r)
                    if (mb[jc0 + t * 16 + 8 * lh + r] == 0) s[t][r] = NEGHUGE;
        }

        // ---- Online softmax: row i is in-lane + one cross-half hop ----
        float mt[8];
        #pragma unroll
        for (int t = 0; t < 8; ++t)
            mt[t] = fmaxf(fmaxf(fmaxf(s[t][0], s[t][1]), fmaxf(s[t][2], s[t][3])),
                          fmaxf(fmaxf(s[t][4], s[t][5]), fmaxf(s[t][6], s[t][7])));
        float mx = fmaxf(fmaxf(fmaxf(mt[0], mt[1]), fmaxf(mt[2], mt[3])),
                         fmaxf(fmaxf(mt[4], mt[5]), fmaxf(mt[6], mt[7])));
        mx = fmaxf(mx, __shfl_xor(mx, 16, 32));

        const float mnew  = fmaxf(rmax, mx);
        const float alpha = EXP2F((rmax - mnew) * LOG2E);
        rmax = mnew;
        const float nrl = -mnew * LOG2E;
        const bool  need = (alpha < 1.0f);

        // P = exp(s - rmax) = exp2(fma(s, log2e, -rmax*log2e))
        #pragma unroll
        for (int t = 0; t < 8; ++t)
            #pragma unroll
            for (int r = 0; r < 8; ++r)
                s[t][r] = EXP2F(__builtin_fmaf(s[t][r], LOG2E, nrl));

        // ---- P^T -> LDS: one b128 store per subtile (8 consecutive j) ----
        #pragma unroll
        for (int t = 0; t < 8; ++t) {
            v8h pk = { (_Float16)s[t][0], (_Float16)s[t][1],
                       (_Float16)s[t][2], (_Float16)s[t][3],
                       (_Float16)s[t][4], (_Float16)s[t][5],
                       (_Float16)s[t][6], (_Float16)s[t][7] };
            *(v8h*)&Pw[(size_t)n * PST_STRIDE + t * 16 + 8 * lh] = pk;
        }
        __builtin_amdgcn_wave_barrier();
        v16h pb[4];   // P^T B-fragments
        #pragma unroll
        for (int c = 0; c < 4; ++c) {
            const _Float16* pp = &Pw[(size_t)n * PST_STRIDE + c * 32 + lh * 16];
            v8h lo = *(const v8h*)pp;
            v8h hi = *(const v8h*)(pp + 8);
            pb[c] = __builtin_shufflevector(lo, hi,
                0,1,2,3,4,5,6,7,8,9,10,11,12,13,14,15);
        }

        // ---- Column-sum of P^T via WMMA (ones as A): per-lane row sum ----
        v8f smv = {};
        #pragma unroll
        for (int c = 0; c < 4; ++c)
            smv = __builtin_amdgcn_wmma_f32_16x16x32_f16(
                false, onesh, false, pb[c], (short)0, smv, false, false);
        const float sm = smv[0];

        if (__any(need)) {
            rsum = rsum * alpha + sm;
            #pragma unroll
            for (int nt = 0; nt < 4; ++nt)
                #pragma unroll
                for (int r = 0; r < 8; ++r)
                    o[nt][r] *= alpha;
        } else {
            rsum += sm;
        }

        // ---- O^T += V^T * P^T : 16 WMMAs ----
        #pragma unroll
        for (int nt = 0; nt < 4; ++nt) {
            #pragma unroll
            for (int c = 0; c < 4; ++c) {
                const _Float16* vp = &Vt[nt * 16 + n][c * 32 + lh * 16];
                v8h lo = *(const v8h*)vp;
                v8h hi = *(const v8h*)(vp + 8);
                v16h va = __builtin_shufflevector(lo, hi,
                    0,1,2,3,4,5,6,7,8,9,10,11,12,13,14,15);
                o[nt] = __builtin_amdgcn_wmma_f32_16x16x32_f16(
                    false, va, false, pb[c], (short)0, o[nt], false, false);
            }
        }
    }

    // ---- Normalize and write out: b128 stores, lane owns one output row ----
    const float inv = 1.0f / rsum;
    float* orow = out + (bh * SEQ + (size_t)iq) * DH;
    #pragma unroll
    for (int nt = 0; nt < 4; ++nt) {
        v4f w0 = { o[nt][0] * inv, o[nt][1] * inv, o[nt][2] * inv, o[nt][3] * inv };
        v4f w1 = { o[nt][4] * inv, o[nt][5] * inv, o[nt][6] * inv, o[nt][7] * inv };
        *(v4f*)(orow + nt * 16 + 8 * lh)     = w0;
        *(v4f*)(orow + nt * 16 + 8 * lh + 4) = w1;
    }
}

extern "C" void kernel_launch(void* const* d_in, const int* in_sizes, int n_in,
                              void* d_out, int out_size, void* d_ws, size_t ws_size,
                              hipStream_t stream) {
    const float*         q    = (const float*)d_in[0];
    const float*         k    = (const float*)d_in[1];
    const float*         v    = (const float*)d_in[2];
    const unsigned char* mask = (const unsigned char*)d_in[3];
    const float*         bias = (const float*)d_in[4];
    float*               out  = (float*)d_out;

    dim3 grid(SEQ / (WAVES * 16), NH, NB); // 16 x 8 x 2 = 256 blocks
    attend_fa_kernel<<<grid, 256, 0, stream>>>(q, k, v, mask, bias, out);
}
